// GraphAttentionLayer_19610820673888
// MI455X (gfx1250) — matrix operand order repaired and checked by reading, not compiled, not run on hardware
//
#include <hip/hip_runtime.h>

#define ALPHA_LRELU 0.2f
#define NEG_BIG -9000000000000000.0f

typedef __attribute__((ext_vector_type(16))) _Float16 v16h;
typedef __attribute__((ext_vector_type(8)))  float    v8f;

// Problem sizes (fixed by the reference)
static constexpr int NROW = 8192;   // items
static constexpr int NCOL = 8192;   // entities
static constexpr int F    = 64;

// ---- workspace layout (units: dwords/floats) ----
static constexpr size_t WS_C1   = 0;                       // 64
static constexpr size_t WS_C2   = 64;                      // 64
static constexpr size_t WS_S1   = 128;                     // 8192
static constexpr size_t WS_S2   = WS_S1 + 8192;            // 8192
static constexpr size_t WS_M    = WS_S2 + 8192;            // 8192  (row max)
static constexpr size_t WS_B    = WS_M  + 8192;            // 262144 (entity f16, WMMA-B swizzled)
static constexpr size_t WS_BITS = WS_B  + 262144;          // 2097152 (adj packed to bits)
static constexpr size_t WS_O    = WS_BITS + 2097152;       // 4*8192*64 (partial O per split)
static constexpr size_t WS_RS   = WS_O  + (size_t)4*8192*64; // 4*8192 (partial row sums)

// K1: c1[k] = sum_f W[k,f]*a[f]; c2[k] = sum_f W[k,f]*a[64+f]
__global__ void k_prep_c(const float* __restrict__ W, const float* __restrict__ a,
                         float* __restrict__ c1, float* __restrict__ c2) {
    int k = threadIdx.x; // 64 threads
    float x1 = 0.f, x2 = 0.f;
#pragma unroll 8
    for (int f = 0; f < F; ++f) {
        float w = W[k * F + f];
        x1 += w * a[f];
        x2 += w * a[F + f];
    }
    c1[k] = x1; c2[k] = x2;
}

// K2: s1[i] = item[i,:]·c1 ; s2[j] = entity[j,:]·c2  (one wave per row)
__global__ __launch_bounds__(256) void k_prep_s(const float* __restrict__ item,
                                                const float* __restrict__ entity,
                                                const float* __restrict__ c1,
                                                const float* __restrict__ c2,
                                                float* __restrict__ s1,
                                                float* __restrict__ s2) {
    int wave = (int)((blockIdx.x * blockDim.x + threadIdx.x) >> 5);
    int lane = threadIdx.x & 31;
    const float* emb; const float* c; float* s; int row;
    if (wave < NROW) { emb = item;   c = c1; s = s1; row = wave; }
    else             { emb = entity; c = c2; s = s2; row = wave - NROW; }
    const float* e = emb + (size_t)row * F;
    float x = e[lane] * c[lane] + e[32 + lane] * c[32 + lane];
#pragma unroll
    for (int off = 16; off; off >>= 1) x += __shfl_down(x, off, 32);
    if (lane == 0) s[row] = x;
}

// K3: swizzle entity -> f16 WMMA-B fragments.
// Fragment (jtile, c): VGPR v, lanes 0-15 hold K=2v,2v+1 (v<4) / K=8+2v,9+2v (v>=4) (col=lane), lanes 16-31 same +8.
// Stored so pass2 lane loads 8 consecutive dwords: wsB[((jt*4+c)*32+lane)*8 + v]
__global__ __launch_bounds__(256) void k_prep_b(const float* __restrict__ entity,
                                                unsigned* __restrict__ wsB) {
    int tid = blockIdx.x * 256 + threadIdx.x;   // 262144 total
    int v    = tid & 7;
    int lane = (tid >> 3) & 31;
    int c    = (tid >> 8) & 3;
    int jt   = tid >> 10;
    int col  = c * 16 + (lane & 15);
    int koff = (lane >> 4) * 8;
    int j0   = jt * 32 + koff + 2 * v + ((v >> 2) << 3);
    float e0 = entity[(size_t)j0 * F + col];
    float e1 = entity[(size_t)(j0 + 1) * F + col];
    unsigned short u0 = __builtin_bit_cast(unsigned short, (_Float16)e0);
    unsigned short u1 = __builtin_bit_cast(unsigned short, (_Float16)e1);
    wsB[tid] = (unsigned)u0 | ((unsigned)u1 << 16);
}

// K4: fused row-max of masked LeakyReLU scores + pack adj>0 into bits (one wave per row).
// Reads the 256MB adj exactly once from HBM; bitmask (8MB) serves pass2 from L2.
__global__ __launch_bounds__(256) void k_rowmax_pack(const int* __restrict__ adj,
                                                     const float* __restrict__ s1,
                                                     const float* __restrict__ s2,
                                                     float* __restrict__ mrow,
                                                     unsigned* __restrict__ bits) {
    int row  = (int)((blockIdx.x * blockDim.x + threadIdx.x) >> 5);
    int lane = threadIdx.x & 31;
    float s1r = s1[row];
    const int* arow = adj + (size_t)row * NCOL;
    float mx = NEG_BIG;
    for (int k = 0; k < 64; ++k) {
        int j = (k << 7) + (lane << 2);
        int4   av = *(const int4*)(arow + j);
        float4 sv = *(const float4*)(s2 + j);
        unsigned nib = 0;
        { float sc = s1r + sv.x; float lr = sc > 0.f ? sc : ALPHA_LRELU * sc;
          float e = av.x > 0 ? lr : NEG_BIG; mx = fmaxf(mx, e); nib |= av.x > 0 ? 1u : 0u; }
        { float sc = s1r + sv.y; float lr = sc > 0.f ? sc : ALPHA_LRELU * sc;
          float e = av.y > 0 ? lr : NEG_BIG; mx = fmaxf(mx, e); nib |= av.y > 0 ? 2u : 0u; }
        { float sc = s1r + sv.z; float lr = sc > 0.f ? sc : ALPHA_LRELU * sc;
          float e = av.z > 0 ? lr : NEG_BIG; mx = fmaxf(mx, e); nib |= av.z > 0 ? 4u : 0u; }
        { float sc = s1r + sv.w; float lr = sc > 0.f ? sc : ALPHA_LRELU * sc;
          float e = av.w > 0 ? lr : NEG_BIG; mx = fmaxf(mx, e); nib |= av.w > 0 ? 8u : 0u; }
        unsigned word = nib << ((lane & 7) << 2);
        word |= __shfl_xor(word, 1, 32);
        word |= __shfl_xor(word, 2, 32);
        word |= __shfl_xor(word, 4, 32);
        if ((lane & 7) == 0) bits[(size_t)row * 256 + (k << 2) + (lane >> 3)] = word;
    }
#pragma unroll
    for (int off = 16; off; off >>= 1) mx = fmaxf(mx, __shfl_down(mx, off, 32));
    if (lane == 0) mrow[row] = mx;
}

// ---- pass2 tile state (software pipeline) ----
struct TileIn {
    unsigned bw;
    float4 sA, sB, sC, sD;
    v16h b0, b1, b2, b3;
};

static __device__ __forceinline__ void load_tile(int w, int lane, int kb,
                                                 const unsigned* __restrict__ brow,
                                                 const float* __restrict__ s2,
                                                 const unsigned* __restrict__ wsB,
                                                 TileIn& T) {
    const int j0 = w << 5;
    T.bw = brow[w];
    T.sA = *(const float4*)(s2 + j0 + kb);
    T.sB = *(const float4*)(s2 + j0 + kb + 4);
    T.sC = *(const float4*)(s2 + j0 + kb + 16);
    T.sD = *(const float4*)(s2 + j0 + kb + 20);
    const unsigned* bp = wsB + ((size_t)w << 10) + ((size_t)lane << 3);
    T.b0 = *(const v16h*)(bp);
    T.b1 = *(const v16h*)(bp + 256);
    T.b2 = *(const v16h*)(bp + 512);
    T.b3 = *(const v16h*)(bp + 768);
}

// K5: fused exp(e - m) + (p @ entity) via v_wmma_f32_16x16x32_f16, + row sums.
// One wave owns a 16-row tile and a 2048-col split (64 K-tiles of 32).
// Software-pipelined: next tile's loads are issued before current tile's VALU+WMMA.
__global__ __launch_bounds__(128) void k_pass2(const unsigned* __restrict__ bits,
                                               const float* __restrict__ s1,
                                               const float* __restrict__ s2,
                                               const float* __restrict__ mrow,
                                               const unsigned* __restrict__ wsB,
                                               float* __restrict__ wsO,
                                               float* __restrict__ wsRS) {
    const int lane  = threadIdx.x & 31;
    const int gw    = (int)((blockIdx.x * blockDim.x + threadIdx.x) >> 5); // 0..2047
    const int itile = gw >> 2;
    const int split = gw & 3;
    const int i0    = itile * 16;
    const int r     = lane & 15;          // A-matrix row owned by this lane
    const int kb    = (lane >> 4) << 3;   // K sub-offset: 0 or 8 per A layout
    const float s1r = s1[i0 + r];
    const float mr  = mrow[i0 + r];
    const unsigned* brow = bits + (size_t)(i0 + r) * 256;

    v8f acc0 = {}, acc1 = {}, acc2 = {}, acc3 = {};
    float rs0 = 0.f, rs1 = 0.f, rs2 = 0.f, rs3 = 0.f;

    const int w0 = split * 64;
    TileIn cur;
    load_tile(w0, lane, kb, brow, s2, wsB, cur);

    for (int t = 0; t < 64; ++t) {
        TileIn nxt;
        if (t < 63) load_tile(w0 + t + 1, lane, kb, brow, s2, wsB, nxt);

        const float sv[16] = { cur.sA.x, cur.sA.y, cur.sA.z, cur.sA.w,
                               cur.sB.x, cur.sB.y, cur.sB.z, cur.sB.w,
                               cur.sC.x, cur.sC.y, cur.sC.z, cur.sC.w,
                               cur.sD.x, cur.sD.y, cur.sD.z, cur.sD.w };
        const unsigned bw = cur.bw;
        float p[16];
        v16h a;
#pragma unroll
        for (int e = 0; e < 16; ++e) {
            // A layout: elements 0..7 -> K = kb+0..7 ; elements 8..15 -> K = kb+16..23
            const int bitidx = kb + (e < 8 ? e : 8 + e);
            const float sc = s1r + sv[e];
            const float lr = sc > 0.f ? sc : ALPHA_LRELU * sc;
            const float ev = ((bw >> bitidx) & 1u) ? lr : NEG_BIG;
            p[e] = __expf(ev - mr);   // masked -> 0 ; empty row -> 1 (matches ref)
            a[e] = (_Float16)p[e];
        }
        // four independent partial-sum chains
        rs0 += p[0] + p[4] + p[8]  + p[12];
        rs1 += p[1] + p[5] + p[9]  + p[13];
        rs2 += p[2] + p[6] + p[10] + p[14];
        rs3 += p[3] + p[7] + p[11] + p[15];

        acc0 = __builtin_amdgcn_wmma_f32_16x16x32_f16(false, a, false, cur.b0, (short)0, acc0, false, false);
        acc1 = __builtin_amdgcn_wmma_f32_16x16x32_f16(false, a, false, cur.b1, (short)0, acc1, false, false);
        acc2 = __builtin_amdgcn_wmma_f32_16x16x32_f16(false, a, false, cur.b2, (short)0, acc2, false, false);
        acc3 = __builtin_amdgcn_wmma_f32_16x16x32_f16(false, a, false, cur.b3, (short)0, acc3, false, false);

        if (t < 63) cur = nxt;
    }

    float rowsum = (rs0 + rs1) + (rs2 + rs3);
    // Full row sum: row r lives in lanes r and r+16 (different K halves)
    rowsum += __shfl_xor(rowsum, 16, 32);
    if (lane < 16) wsRS[(size_t)split * NROW + i0 + lane] = rowsum;

    // D layout: VGPR t -> row t (lanes 0-15) / row t+8 (lanes 16-31); col = lane&15
    const int mro = (lane >> 4) << 3;
    const int col = lane & 15;
#pragma unroll
    for (int t2 = 0; t2 < 8; ++t2) {
        const size_t ro = ((size_t)split * NROW + i0 + mro + t2) * F + col;
        wsO[ro +  0] = acc0[t2];
        wsO[ro + 16] = acc1[t2];
        wsO[ro + 32] = acc2[t2];
        wsO[ro + 48] = acc3[t2];
    }
}

// K6: combine splits, normalize, +item, ELU
__global__ __launch_bounds__(256) void k_final(const float* __restrict__ wsO,
                                               const float* __restrict__ wsRS,
                                               const float* __restrict__ item,
                                               float* __restrict__ out) {
    int tid = blockIdx.x * 256 + threadIdx.x;   // 524288
    int i = tid >> 6;
    int f = tid & 63;
    float o = 0.f, s = 0.f;
#pragma unroll
    for (int sp = 0; sp < 4; ++sp) {
        o += wsO[((size_t)sp * NROW + i) * F + f];
        s += wsRS[(size_t)sp * NROW + i];
    }
    float v = o / s + item[tid];
    out[tid] = v > 0.f ? v : expm1f(v);
}

extern "C" void kernel_launch(void* const* d_in, const int* in_sizes, int n_in,
                              void* d_out, int out_size, void* d_ws, size_t ws_size,
                              hipStream_t stream) {
    (void)in_sizes; (void)n_in; (void)out_size; (void)ws_size;
    const float* item   = (const float*)d_in[0];
    const float* entity = (const float*)d_in[1];
    const int*   adj    = (const int*)d_in[2];
    const float* W      = (const float*)d_in[3];
    const float* a      = (const float*)d_in[4];

    float* ws = (float*)d_ws;
    float*    c1   = ws + WS_C1;
    float*    c2   = ws + WS_C2;
    float*    s1   = ws + WS_S1;
    float*    s2   = ws + WS_S2;
    float*    mrow = ws + WS_M;
    unsigned* wsB  = (unsigned*)(ws + WS_B);
    unsigned* bits = (unsigned*)(ws + WS_BITS);
    float*    wsO  = ws + WS_O;
    float*    wsRS = ws + WS_RS;
    float*    out  = (float*)d_out;

    k_prep_c<<<1, 64, 0, stream>>>(W, a, c1, c2);
    k_prep_s<<<2048, 256, 0, stream>>>(item, entity, c1, c2, s1, s2);
    k_prep_b<<<1024, 256, 0, stream>>>(entity, wsB);
    k_rowmax_pack<<<1024, 256, 0, stream>>>(adj, s1, s2, mrow, bits);
    k_pass2<<<512, 128, 0, stream>>>(bits, s1, s2, mrow, wsB, wsO, wsRS);
    k_final<<<2048, 256, 0, stream>>>(wsO, wsRS, item, out);
}
